// DecoderLayer_9371618640202
// MI455X (gfx1250) — matrix operand order repaired and checked by reading, not compiled, not run on hardware
//
#include <hip/hip_runtime.h>
#include <hip/hip_bf16.h>

// ---------- types ----------
typedef __bf16 v16bf  __attribute__((ext_vector_type(16)));
typedef __bf16 bf16x8 __attribute__((ext_vector_type(8)));
typedef float  v8f    __attribute__((ext_vector_type(8)));

#define S_LEN 2048
#define E_DIM 1024
#define NHEAD 16
#define DHEAD 64
#define BATCH 2
#define NTOK  (BATCH * S_LEN)   // 4096 rows for all big GEMMs

__device__ __forceinline__ float neg_inf() { return -__builtin_inff(); }

// Assemble a v16bf fragment from two 16-byte LDS chunks (ds_load_b128 x2).
__device__ __forceinline__ v16bf make_frag(const __bf16* p0, const __bf16* p1) {
  bf16x8 lo = *(const bf16x8*)p0;
  bf16x8 hi = *(const bf16x8*)p1;
  return __builtin_shufflevector(lo, hi, 0,1,2,3,4,5,6,7,8,9,10,11,12,13,14,15);
}

// ---------- prep kernels ----------
__global__ void f32_to_bf16_kernel(const float* __restrict__ X,
                                   __bf16* __restrict__ Y, long n) {
  long i = (long)blockIdx.x * blockDim.x + threadIdx.x;
  if (i < n) Y[i] = (__bf16)X[i];
}

// W is [K][N] fp32 (y = x@W), emit Wt as [N][K] bf16 so GEMM B-tiles are K-contiguous.
__global__ void transpose_to_bf16_kernel(const float* __restrict__ W,
                                         __bf16* __restrict__ Wt, int K, int N) {
  long i = (long)blockIdx.x * blockDim.x + threadIdx.x;
  if (i < (long)K * N) {
    int k = (int)(i / N), n = (int)(i % N);
    Wt[(long)n * K + k] = (__bf16)W[i];
  }
}

// ---------- tiled WMMA GEMM: C[M,N] = A[M,K](bf16) * Bt[N,K](bf16)^T ----------
// 256 threads = 8 waves; tile 128x128x32; wave computes 32x64 (2x4 WMMA tiles).
// Ping-pong LDS double buffering: global loads for tile k+1 overlap WMMA on tile k.
// VMODE stores bf16 output in [B,H,DHEAD,S] layout (for attention V).
template <bool HAS_RES, bool HAS_F32, bool HAS_BF16, bool RELU, bool VMODE>
__global__ __launch_bounds__(256) void gemm_bf16_wmma(
    const __bf16* __restrict__ A, const __bf16* __restrict__ Bt,
    const float* __restrict__ bias, const float* __restrict__ resid,
    float* __restrict__ Cf, __bf16* __restrict__ Cb,
    int M, int N, int K)
{
  __shared__ __align__(16) __bf16 sA[2][128 * 32];
  __shared__ __align__(16) __bf16 sB[2][128 * 32];

  const int tid  = threadIdx.x;
  const int lane = tid & 31, wid = tid >> 5;
  const int half = lane >> 4, l16 = lane & 15;
  const int wm = wid >> 1, wn = wid & 1;           // 4x2 wave grid
  const long rowA0 = (long)blockIdx.y * 128;
  const long rowB0 = (long)blockIdx.x * 128;

  // per-thread cooperative-load coordinates: two 8-elem chunks per tile
  const int r0 = tid >> 2;              // 0..63
  const int r1 = r0 + 64;               // 64..127
  const int c0 = (tid & 3) * 8;         // 0,8,16,24

  v8f acc[2][4];
  #pragma unroll
  for (int i = 0; i < 2; i++)
    #pragma unroll
    for (int j = 0; j < 4; j++)
      #pragma unroll
      for (int e = 0; e < 8; e++) acc[i][j][e] = 0.f;

  bf16x8 ra0, ra1, rb0, rb1;
  auto gload = [&](int kt) {
    const long kk = (long)kt * 32;
    ra0 = *(const bf16x8*)&A [(rowA0 + r0) * K + kk + c0];
    ra1 = *(const bf16x8*)&A [(rowA0 + r1) * K + kk + c0];
    rb0 = *(const bf16x8*)&Bt[(rowB0 + r0) * K + kk + c0];
    rb1 = *(const bf16x8*)&Bt[(rowB0 + r1) * K + kk + c0];
  };
  auto sstore = [&](int buf) {
    *(bf16x8*)&sA[buf][r0 * 32 + c0] = ra0;
    *(bf16x8*)&sA[buf][r1 * 32 + c0] = ra1;
    *(bf16x8*)&sB[buf][r0 * 32 + c0] = rb0;
    *(bf16x8*)&sB[buf][r1 * 32 + c0] = rb1;
  };

  const int nk = K >> 5;
  gload(0);
  sstore(0);
  __syncthreads();

  for (int kt = 0; kt < nk; kt++) {
    const int cur = kt & 1;
    if (kt + 1 < nk) gload(kt + 1);    // global loads in flight during WMMA

    v16bf af[2], bfr[4];
    #pragma unroll
    for (int i = 0; i < 2; i++) {
      const __bf16* rp = &sA[cur][(wm * 32 + i * 16 + l16) * 32];
      af[i] = make_frag(rp + half * 16, rp + half * 16 + 8);
    }
    #pragma unroll
    for (int j = 0; j < 4; j++) {
      const __bf16* rp = &sB[cur][(wn * 64 + j * 16 + l16) * 32];
      bfr[j] = make_frag(rp + half * 16, rp + half * 16 + 8);
    }
    #pragma unroll
    for (int i = 0; i < 2; i++)
      #pragma unroll
      for (int j = 0; j < 4; j++)
        acc[i][j] = __builtin_amdgcn_wmma_f32_16x16x32_bf16(
            false, af[i], false, bfr[j], (short)0, acc[i][j], false, false);

    if (kt + 1 < nk) sstore(cur ^ 1);  // fill the other buffer
    __syncthreads();
  }

  // Epilogue: C-layout element r -> row half*8+r, col = lane%16 within tile.
  #pragma unroll
  for (int i = 0; i < 2; i++) {
    #pragma unroll
    for (int j = 0; j < 4; j++) {
      const int gcol = (int)rowB0 + wn * 64 + j * 16 + l16;
      const float bv = bias[gcol];
      #pragma unroll
      for (int r = 0; r < 8; r++) {
        const long grow = rowA0 + wm * 32 + i * 16 + half * 8 + r;
        float v = acc[i][j][r] + bv;
        if constexpr (HAS_RES) v += resid[grow * N + gcol];
        if constexpr (RELU)    v = v > 0.f ? v : 0.f;
        if constexpr (HAS_F32) Cf[grow * N + gcol] = v;
        if constexpr (HAS_BF16) {
          if constexpr (VMODE) {       // V store: [b][h][d][s]
            const int b = (int)(grow / S_LEN), s = (int)(grow % S_LEN);
            const int h = gcol / DHEAD, d = gcol % DHEAD;
            Cb[((long)((b * NHEAD + h) * DHEAD + d)) * S_LEN + s] = (__bf16)v;
          } else {
            Cb[grow * N + gcol] = (__bf16)v;
          }
        }
      }
    }
  }
}

// ---------- flash attention (WMMA QK^T + PV, online softmax) ----------
// 128 threads = 4 waves; block = 64 query rows of one (b,h); key blocks of 64.
__global__ __launch_bounds__(128) void flash_attn_wmma(
    const __bf16* __restrict__ Q,    // [NTOK, E_DIM], head h at col h*64
    const __bf16* __restrict__ Kmat, // [NTOK, E_DIM]
    const __bf16* __restrict__ Vt,   // [(b*H+h)*64+d][S]
    const unsigned char* __restrict__ mask, // [B, S] over key positions
    __bf16* __restrict__ O,          // [NTOK, E_DIM]
    int causal)
{
  __shared__ __align__(16) __bf16 sQ[64 * 64];
  __shared__ __align__(16) __bf16 sK[64 * 64];
  __shared__ __align__(16) __bf16 sV[64 * 64];     // [d][key]
  __shared__ __align__(16) __bf16 sP[4 * 16 * 64]; // per-wave P tile

  const int tid  = threadIdx.x;
  const int lane = tid & 31, w = tid >> 5;
  const int half = lane >> 4, l16 = lane & 15;
  const int qt = blockIdx.x, bh = blockIdx.y;
  const int b = bh / NHEAD, h = bh % NHEAD;

  #pragma unroll
  for (int rr = 0; rr < 4; rr++) {                 // load Q tile 64x64
    int g = tid + rr * 128;
    int r = g >> 3, c = (g & 7) * 8;
    *(bf16x8*)&sQ[r * 64 + c] =
        *(const bf16x8*)&Q[((long)(b * S_LEN + qt * 64 + r)) * E_DIM + h * DHEAD + c];
  }

  float mrow[8], lrow[8];
  v8f oacc[4];
  #pragma unroll
  for (int r = 0; r < 8; r++) { mrow[r] = neg_inf(); lrow[r] = 0.f; }
  #pragma unroll
  for (int j = 0; j < 4; j++)
    #pragma unroll
    for (int e = 0; e < 8; e++) oacc[j][e] = 0.f;

  const int kbEnd = causal ? qt : (S_LEN / 64 - 1);
  for (int kb = 0; kb <= kbEnd; kb++) {
    __syncthreads();
    #pragma unroll
    for (int rr = 0; rr < 4; rr++) {
      int g = tid + rr * 128;
      int r = g >> 3, c = (g & 7) * 8;
      *(bf16x8*)&sK[r * 64 + c] =
          *(const bf16x8*)&Kmat[((long)(b * S_LEN + kb * 64 + r)) * E_DIM + h * DHEAD + c];
      *(bf16x8*)&sV[r * 64 + c] =
          *(const bf16x8*)&Vt[((long)((b * NHEAD + h) * DHEAD + r)) * S_LEN + kb * 64 + c];
    }
    __syncthreads();

    // prefetch next K/V tiles into cache while we crunch this block
    if (kb < kbEnd) {
      const int pr = tid >> 1, pc = (tid & 1) * 32;
      __builtin_prefetch(
          &Kmat[((long)(b * S_LEN + (kb + 1) * 64 + pr)) * E_DIM + h * DHEAD + pc], 0, 1);
      __builtin_prefetch(
          &Vt[((long)((b * NHEAD + h) * DHEAD + pr)) * S_LEN + (kb + 1) * 64 + pc], 0, 1);
    }

    // scores: wave computes 16 rows x 64 key cols
    v8f sc[4];
    #pragma unroll
    for (int j = 0; j < 4; j++)
      #pragma unroll
      for (int e = 0; e < 8; e++) sc[j][e] = 0.f;
    #pragma unroll
    for (int ks = 0; ks < 2; ks++) {
      const __bf16* qp = &sQ[(w * 16 + l16) * 64 + ks * 32];
      v16bf aq = make_frag(qp + half * 16, qp + half * 16 + 8);
      #pragma unroll
      for (int j = 0; j < 4; j++) {
        const __bf16* kp = &sK[(j * 16 + l16) * 64 + ks * 32];
        v16bf bk = make_frag(kp + half * 16, kp + half * 16 + 8);
        sc[j] = __builtin_amdgcn_wmma_f32_16x16x32_bf16(
            false, aq, false, bk, (short)0, sc[j], false, false);
      }
    }

    // mask -> -inf, then scale by 1/sqrt(64)  (reference order: mask then scale)
    #pragma unroll
    for (int j = 0; j < 4; j++) {
      int kcol = kb * 64 + j * 16 + l16;
      bool ok = mask[b * S_LEN + kcol] != 0;
      #pragma unroll
      for (int r = 0; r < 8; r++) {
        int qrow = qt * 64 + w * 16 + half * 8 + r;
        bool dead = !ok || (causal && kcol > qrow);
        sc[j][r] = dead ? neg_inf() : sc[j][r] * 0.125f;
      }
    }

    // online softmax; rows live across 16-lane groups (fixed half)
    #pragma unroll
    for (int r = 0; r < 8; r++) {
      float pm = sc[0][r];
      #pragma unroll
      for (int j = 1; j < 4; j++) pm = fmaxf(pm, sc[j][r]);
      #pragma unroll
      for (int d = 1; d < 16; d <<= 1) pm = fmaxf(pm, __shfl_xor(pm, d, 32));
      float mn   = fmaxf(mrow[r], pm);
      bool  live = mn > neg_inf();
      float corr = (mrow[r] == neg_inf()) ? 0.f : __expf(mrow[r] - mn);
      float psum = 0.f;
      #pragma unroll
      for (int j = 0; j < 4; j++) {
        float pv = live ? __expf(sc[j][r] - mn) : 0.f;
        psum += pv;
        sP[w * 1024 + (half * 8 + r) * 64 + j * 16 + l16] = (__bf16)pv;
      }
      #pragma unroll
      for (int d = 1; d < 16; d <<= 1) psum += __shfl_xor(psum, d, 32);
      lrow[r] = lrow[r] * corr + psum;
      mrow[r] = mn;
      #pragma unroll
      for (int j = 0; j < 4; j++) oacc[j][r] *= corr;
    }

    // P @ V  (sP is private to this wave; compiler orders ds store->load)
    #pragma unroll
    for (int ks = 0; ks < 2; ks++) {
      const __bf16* pp = &sP[w * 1024 + l16 * 64 + ks * 32];
      v16bf ap = make_frag(pp + half * 16, pp + half * 16 + 8);
      #pragma unroll
      for (int j = 0; j < 4; j++) {
        const __bf16* vp = &sV[(j * 16 + l16) * 64 + ks * 32];
        v16bf bv = make_frag(vp + half * 16, vp + half * 16 + 8);
        oacc[j] = __builtin_amdgcn_wmma_f32_16x16x32_bf16(
            false, ap, false, bv, (short)0, oacc[j], false, false);
      }
    }
  }

  #pragma unroll
  for (int j = 0; j < 4; j++)
    #pragma unroll
    for (int r = 0; r < 8; r++) {
      float v = (lrow[r] > 0.f) ? oacc[j][r] / lrow[r] : 0.f;
      long row = (long)(b * S_LEN + qt * 64 + w * 16 + half * 8 + r);
      O[row * E_DIM + h * DHEAD + j * 16 + l16] = (__bf16)v;
    }
}

// ---------- layer norm over last dim (E_DIM), optional f32 + bf16 outputs ----------
__global__ __launch_bounds__(256) void layernorm_kernel(
    const float* __restrict__ X, const float* __restrict__ gam,
    const float* __restrict__ bet, float* __restrict__ Yf,
    __bf16* __restrict__ Yb)
{
  __shared__ float s1[256], s2[256];
  const int tid = threadIdx.x;
  const long row = blockIdx.x;
  const float4 x4 = ((const float4*)(X + row * E_DIM))[tid];
  float sum = x4.x + x4.y + x4.z + x4.w;
  float sq  = x4.x * x4.x + x4.y * x4.y + x4.z * x4.z + x4.w * x4.w;
  s1[tid] = sum; s2[tid] = sq;
  __syncthreads();
  for (int off = 128; off > 0; off >>= 1) {
    if (tid < off) { s1[tid] += s1[tid + off]; s2[tid] += s2[tid + off]; }
    __syncthreads();
  }
  float mean = s1[0] * (1.f / E_DIM);
  float var  = s2[0] * (1.f / E_DIM) - mean * mean;
  float rstd = rsqrtf(var + 1e-12f);
  float xv[4] = {x4.x, x4.y, x4.z, x4.w};
  #pragma unroll
  for (int u = 0; u < 4; u++) {
    int c = tid * 4 + u;
    float y = gam[c] * (xv[u] - mean) * rstd + bet[c];
    if (Yf) Yf[row * E_DIM + c] = y;
    if (Yb) Yb[row * E_DIM + c] = (__bf16)y;
  }
}

// ---------- host orchestration ----------
extern "C" void kernel_launch(void* const* d_in, const int* in_sizes, int n_in,
                              void* d_out, int out_size, void* d_ws, size_t ws_size,
                              hipStream_t stream) {
  (void)in_sizes; (void)n_in; (void)out_size; (void)ws_size;
  const float* input = (const float*)d_in[0];
  const float* enc   = (const float*)d_in[1];
  const unsigned char* src_mask = (const unsigned char*)d_in[2];
  const unsigned char* trg_mask = (const unsigned char*)d_in[3];
  const float* sa_wq = (const float*)d_in[4];  const float* sa_bq = (const float*)d_in[5];
  const float* sa_wk = (const float*)d_in[6];  const float* sa_bk = (const float*)d_in[7];
  const float* sa_wv = (const float*)d_in[8];  const float* sa_bv = (const float*)d_in[9];
  const float* sa_wo = (const float*)d_in[10]; const float* sa_bo = (const float*)d_in[11];
  const float* ca_wq = (const float*)d_in[12]; const float* ca_bq = (const float*)d_in[13];
  const float* ca_wk = (const float*)d_in[14]; const float* ca_bk = (const float*)d_in[15];
  const float* ca_wv = (const float*)d_in[16]; const float* ca_bv = (const float*)d_in[17];
  const float* ca_wo = (const float*)d_in[18]; const float* ca_bo = (const float*)d_in[19];
  const float* ffn_w1 = (const float*)d_in[20]; const float* ffn_b1 = (const float*)d_in[21];
  const float* ffn_w2 = (const float*)d_in[22]; const float* ffn_b2 = (const float*)d_in[23];
  const float* ln1_g = (const float*)d_in[24]; const float* ln1_b = (const float*)d_in[25];
  const float* ln2_g = (const float*)d_in[26]; const float* ln2_b = (const float*)d_in[27];
  const float* ln3_g = (const float*)d_in[28]; const float* ln3_b = (const float*)d_in[29];

  char* ws = (char*)d_ws;
  size_t off = 0;
  auto alloc = [&](size_t bytes) -> char* {
    char* p = ws + off;
    off += (bytes + 255) & ~(size_t)255;
    return p;
  };
  const size_t EE = (size_t)E_DIM * E_DIM;
  const size_t TOKE = (size_t)NTOK * E_DIM;

  __bf16* wqt_sa = (__bf16*)alloc(EE * 2);
  __bf16* wkt_sa = (__bf16*)alloc(EE * 2);
  __bf16* wvt_sa = (__bf16*)alloc(EE * 2);
  __bf16* wot_sa = (__bf16*)alloc(EE * 2);
  __bf16* wqt_ca = (__bf16*)alloc(EE * 2);
  __bf16* wkt_ca = (__bf16*)alloc(EE * 2);
  __bf16* wvt_ca = (__bf16*)alloc(EE * 2);
  __bf16* wot_ca = (__bf16*)alloc(EE * 2);
  __bf16* w1t    = (__bf16*)alloc(EE * 4 * 2);   // [4096][1024]
  __bf16* w2t    = (__bf16*)alloc(EE * 4 * 2);   // [1024][4096]
  __bf16* x_bf   = (__bf16*)alloc(TOKE * 2);
  __bf16* enc_bf = (__bf16*)alloc(TOKE * 2);
  __bf16* q_bf   = (__bf16*)alloc(TOKE * 2);
  __bf16* k_bf   = (__bf16*)alloc(TOKE * 2);
  __bf16* vt_bf  = (__bf16*)alloc(TOKE * 2);
  __bf16* attn_bf= (__bf16*)alloc(TOKE * 2);
  float*  y_f32  = (float*)alloc(TOKE * 4);      // reused across the 3 residual sums
  float*  x1_f32 = (float*)alloc(TOKE * 4);
  __bf16* x1_bf  = (__bf16*)alloc(TOKE * 2);
  float*  x2_f32 = (float*)alloc(TOKE * 4);
  __bf16* x2_bf  = (__bf16*)alloc(TOKE * 2);
  __bf16* h_bf   = (__bf16*)alloc((size_t)NTOK * 4 * E_DIM * 2);

  const int TB = 256;
  dim3 gT((unsigned)((EE + TB - 1) / TB));
  dim3 gT4((unsigned)((EE * 4 + TB - 1) / TB));
  dim3 gC((unsigned)((TOKE + TB - 1) / TB));

  // weight prep (fp32 [K][N] -> bf16 [N][K])
  transpose_to_bf16_kernel<<<gT, TB, 0, stream>>>(sa_wq, wqt_sa, E_DIM, E_DIM);
  transpose_to_bf16_kernel<<<gT, TB, 0, stream>>>(sa_wk, wkt_sa, E_DIM, E_DIM);
  transpose_to_bf16_kernel<<<gT, TB, 0, stream>>>(sa_wv, wvt_sa, E_DIM, E_DIM);
  transpose_to_bf16_kernel<<<gT, TB, 0, stream>>>(sa_wo, wot_sa, E_DIM, E_DIM);
  transpose_to_bf16_kernel<<<gT, TB, 0, stream>>>(ca_wq, wqt_ca, E_DIM, E_DIM);
  transpose_to_bf16_kernel<<<gT, TB, 0, stream>>>(ca_wk, wkt_ca, E_DIM, E_DIM);
  transpose_to_bf16_kernel<<<gT, TB, 0, stream>>>(ca_wv, wvt_ca, E_DIM, E_DIM);
  transpose_to_bf16_kernel<<<gT, TB, 0, stream>>>(ca_wo, wot_ca, E_DIM, E_DIM);
  transpose_to_bf16_kernel<<<gT4, TB, 0, stream>>>(ffn_w1, w1t, E_DIM, 4 * E_DIM);
  transpose_to_bf16_kernel<<<gT4, TB, 0, stream>>>(ffn_w2, w2t, 4 * E_DIM, E_DIM);
  f32_to_bf16_kernel<<<gC, TB, 0, stream>>>(input, x_bf, (long)TOKE);
  f32_to_bf16_kernel<<<gC, TB, 0, stream>>>(enc, enc_bf, (long)TOKE);

  dim3 gemmE(E_DIM / 128, NTOK / 128);        // (8, 32)
  dim3 gemmH(4 * E_DIM / 128, NTOK / 128);    // (32, 32)
  dim3 flashG(S_LEN / 64, BATCH * NHEAD);     // (32, 32)

  // template configs: <HAS_RES, HAS_F32, HAS_BF16, RELU, VMODE>
  // ---- self attention ----
  gemm_bf16_wmma<false, false, true, false, false><<<gemmE, 256, 0, stream>>>(
      x_bf, wqt_sa, sa_bq, nullptr, nullptr, q_bf, NTOK, E_DIM, E_DIM);
  gemm_bf16_wmma<false, false, true, false, false><<<gemmE, 256, 0, stream>>>(
      x_bf, wkt_sa, sa_bk, nullptr, nullptr, k_bf, NTOK, E_DIM, E_DIM);
  gemm_bf16_wmma<false, false, true, false, true><<<gemmE, 256, 0, stream>>>(
      x_bf, wvt_sa, sa_bv, nullptr, nullptr, vt_bf, NTOK, E_DIM, E_DIM);
  flash_attn_wmma<<<flashG, 128, 0, stream>>>(q_bf, k_bf, vt_bf, trg_mask, attn_bf, 1);
  gemm_bf16_wmma<true, true, false, false, false><<<gemmE, 256, 0, stream>>>(
      attn_bf, wot_sa, sa_bo, input, y_f32, nullptr, NTOK, E_DIM, E_DIM);
  layernorm_kernel<<<NTOK, 256, 0, stream>>>(y_f32, ln1_g, ln1_b, x1_f32, x1_bf);

  // ---- cross attention ----
  gemm_bf16_wmma<false, false, true, false, false><<<gemmE, 256, 0, stream>>>(
      x1_bf, wqt_ca, ca_bq, nullptr, nullptr, q_bf, NTOK, E_DIM, E_DIM);
  gemm_bf16_wmma<false, false, true, false, false><<<gemmE, 256, 0, stream>>>(
      enc_bf, wkt_ca, ca_bk, nullptr, nullptr, k_bf, NTOK, E_DIM, E_DIM);
  gemm_bf16_wmma<false, false, true, false, true><<<gemmE, 256, 0, stream>>>(
      enc_bf, wvt_ca, ca_bv, nullptr, nullptr, vt_bf, NTOK, E_DIM, E_DIM);
  flash_attn_wmma<<<flashG, 128, 0, stream>>>(q_bf, k_bf, vt_bf, src_mask, attn_bf, 0);
  gemm_bf16_wmma<true, true, false, false, false><<<gemmE, 256, 0, stream>>>(
      attn_bf, wot_ca, ca_bo, x1_f32, y_f32, nullptr, NTOK, E_DIM, E_DIM);
  layernorm_kernel<<<NTOK, 256, 0, stream>>>(y_f32, ln2_g, ln2_b, x2_f32, x2_bf);

  // ---- FFN ----
  gemm_bf16_wmma<false, false, true, true, false><<<gemmH, 256, 0, stream>>>(
      x2_bf, w1t, ffn_b1, nullptr, nullptr, h_bf, NTOK, 4 * E_DIM, E_DIM);
  gemm_bf16_wmma<true, true, false, false, false><<<gemmE, 256, 0, stream>>>(
      h_bf, w2t, ffn_b2, x2_f32, y_f32, nullptr, NTOK, E_DIM, 4 * E_DIM);
  layernorm_kernel<<<NTOK, 256, 0, stream>>>(y_f32, ln3_g, ln3_b, (float*)d_out, nullptr);
}
